// FusedLowRankMoE_Reordered_34961033789984
// MI455X (gfx1250) — matrix operand order repaired and checked by reading, not compiled, not run on hardware
//
#include <hip/hip_runtime.h>

typedef __bf16 bf16_t;
typedef __attribute__((ext_vector_type(16))) __bf16 v16bf;
typedef __attribute__((ext_vector_type(8)))  __bf16 v8bf;
typedef __attribute__((ext_vector_type(8)))  float  v8f;

#define D_MODEL  1024
#define HK       32      // NUM_HEADS * TOP_K
#define D_LATENT 64
#define D_HIDDEN 128
#define TILE_M   16

// fragment-ordered bf16 weight images in workspace
#define WU_ELEMS (D_MODEL * D_HIDDEN)          // 131072
#define WV_ELEMS (D_HIDDEN * D_MODEL)          // 131072
#define W1_ELEMS (D_LATENT * D_HIDDEN)         // 8192
#define WS_ELEMS (WU_ELEMS + WV_ELEMS + W1_ELEMS)

__device__ __forceinline__ float gelu_sig(float v) {
    // v * sigmoid(1.702 v) == v / (1 + exp(-1.702 v))
    return v / (1.0f + __expf(-1.702f * v));
}

__device__ __forceinline__ v16bf mk16(v8bf lo, v8bf hi) {
    return __builtin_shufflevector(lo, hi, 0,1,2,3,4,5,6,7,8,9,10,11,12,13,14,15);
}

__device__ __forceinline__ v16bf cvt16(float4 p0, float4 p1, float4 p2, float4 p3) {
    v16bf a;
    a[0]=(bf16_t)p0.x;  a[1]=(bf16_t)p0.y;  a[2]=(bf16_t)p0.z;  a[3]=(bf16_t)p0.w;
    a[4]=(bf16_t)p1.x;  a[5]=(bf16_t)p1.y;  a[6]=(bf16_t)p1.z;  a[7]=(bf16_t)p1.w;
    a[8]=(bf16_t)p2.x;  a[9]=(bf16_t)p2.y;  a[10]=(bf16_t)p2.z; a[11]=(bf16_t)p2.w;
    a[12]=(bf16_t)p3.x; a[13]=(bf16_t)p3.y; a[14]=(bf16_t)p3.z; a[15]=(bf16_t)p3.w;
    return a;
}

// ---------------------------------------------------------------------------
// Prepass: convert W_u / W_v / W1 (f32) into bf16, pre-swizzled into WMMA
// B-fragment order:  frag[c][t][n][half][e] = W[(c*32 + (e&7) + 8*half + 16*(e>>3))][t*16 + n]
// so a lane's 16 B-fragment elements are one contiguous 32-byte run.
// ---------------------------------------------------------------------------
__global__ void convert_weights_kernel(const float* __restrict__ Wu,
                                       const float* __restrict__ Wv,
                                       const float* __restrict__ W1,
                                       bf16_t* __restrict__ ws) {
    const int i = blockIdx.x * 256 + threadIdx.x;
    if (i < WU_ELEMS) {
        const int e = i & 15, half = (i >> 4) & 1, n = (i >> 5) & 15;
        const int t = (i >> 9) & 7, c = i >> 12;                 // 8 tiles, 32 chunks
        ws[i] = (bf16_t)Wu[(c * 32 + (e & 7) + 8 * half + 16 * (e >> 3)) * D_HIDDEN + t * 16 + n];
    } else if (i < WU_ELEMS + WV_ELEMS) {
        const int j = i - WU_ELEMS;
        const int e = j & 15, half = (j >> 4) & 1, n = (j >> 5) & 15;
        const int t = (j >> 9) & 63, c = j >> 15;                // 64 tiles, 4 chunks
        ws[i] = (bf16_t)Wv[(c * 32 + (e & 7) + 8 * half + 16 * (e >> 3)) * D_MODEL + t * 16 + n];
    } else if (i < WS_ELEMS) {
        const int j = i - (WU_ELEMS + WV_ELEMS);
        const int e = j & 15, half = (j >> 4) & 1, n = (j >> 5) & 15;
        const int t = (j >> 9) & 7, c = j >> 12;                 // 8 tiles, 2 chunks
        ws[i] = (bf16_t)W1[(c * 32 + (e & 7) + 8 * half + 16 * (e >> 3)) * D_HIDDEN + t * 16 + n];
    }
}

// ---------------------------------------------------------------------------
// Main fused kernel: one 4-wave workgroup per 16-token tile.
// ---------------------------------------------------------------------------
__launch_bounds__(128)
__global__ void fused_lowrank_moe_kernel(const float* __restrict__ x,
                                         const float* __restrict__ scores,
                                         const int*   __restrict__ eidx,
                                         const float* __restrict__ lat,
                                         const bf16_t* __restrict__ wuf,
                                         const bf16_t* __restrict__ wvf,
                                         const bf16_t* __restrict__ w1f,
                                         float* __restrict__ out) {
    __shared__ alignas(16) bf16_t x_lds[TILE_M * D_MODEL];      // 32 KB
    __shared__ alignas(16) bf16_t w1_lds[W1_ELEMS];             // 16 KB (fragment order)
    __shared__ alignas(16) float  xp_lds[TILE_M][D_HIDDEN];     //  8 KB
    __shared__ alignas(16) float  h_ws[4][TILE_M][D_HIDDEN];    // 32 KB (per-wave)
    __shared__ float  act_ws[4][TILE_M];
    __shared__ alignas(16) bf16_t accA[TILE_M * D_HIDDEN];      //  4 KB

    const int tid  = threadIdx.x;
    const int wave = tid >> 5;
    const int lane = tid & 31;
    const int lm   = lane & 15;
    const int half = lane >> 4;
    const int n0   = blockIdx.x * TILE_M;
    const int fofs = lm * 32 + half * 16;      // lane offset inside a fragment block

    // ---- stage x tile (f32 -> bf16) and fragment-ordered W1 into LDS ----
    {
        const float4* xg = (const float4*)(x + (size_t)n0 * D_MODEL);
        for (int i = tid; i < TILE_M * D_MODEL / 4; i += 128) {
            float4 v = xg[i];
            x_lds[4 * i + 0] = (bf16_t)v.x;
            x_lds[4 * i + 1] = (bf16_t)v.y;
            x_lds[4 * i + 2] = (bf16_t)v.z;
            x_lds[4 * i + 3] = (bf16_t)v.w;
        }
        const float4* wg = (const float4*)w1f;   // raw 16 KB copy, 8 bf16 per float4
        float4* wl = (float4*)w1_lds;
        for (int i = tid; i < W1_ELEMS / 8; i += 128) wl[i] = wg[i];
    }
    __syncthreads();

    // ---- Phase A: x_proj[16][128] = x_tile @ W_u ----
    {
        v8f xc0 = {}; v8f xc1 = {};
        const int t0 = wave * 2, t1 = wave * 2 + 1;
        const bf16_t* xrow = x_lds + lm * D_MODEL + 8 * half;
        for (int c = 0; c < D_MODEL / 32; ++c) {
            v16bf a = mk16(*(const v8bf*)(xrow + c * 32),
                           *(const v8bf*)(xrow + c * 32 + 16));
            const bf16_t* bu0 = wuf + (size_t)(c * 8 + t0) * 512 + fofs;
            const bf16_t* bu1 = wuf + (size_t)(c * 8 + t1) * 512 + fofs;
            v16bf b0 = mk16(*(const v8bf*)bu0, *(const v8bf*)(bu0 + 8));
            v16bf b1 = mk16(*(const v8bf*)bu1, *(const v8bf*)(bu1 + 8));
            xc0 = __builtin_amdgcn_wmma_f32_16x16x32_bf16(false, a, false, b0, (short)0, xc0, false, false);
            xc1 = __builtin_amdgcn_wmma_f32_16x16x32_bf16(false, a, false, b1, (short)0, xc1, false, false);
        }
        #pragma unroll
        for (int r = 0; r < 8; ++r) {
            xp_lds[r + 8 * half][t0 * 16 + lm] = xc0[r];
            xp_lds[r + 8 * half][t1 * 16 + lm] = xc1[r];
        }
    }
    __syncthreads();

    // ---- Phase B: per (head,k) group of the 16 tokens ----
    v8f acc[8];
    {
        v8f z = {};
        #pragma unroll
        for (int t = 0; t < 8; ++t) acc[t] = z;
    }

    for (int gi = 0; gi < 8; ++gi) {
        const int g = wave * 8 + gi;

        // gather latent row for token (n0+lm), expert slot g, into A-fragment order
        const int idx = eidx[(size_t)(n0 + lm) * HK + g];
        const float4* lp = (const float4*)(lat + (size_t)idx * D_LATENT);
        v16bf a0 = cvt16(lp[2 * half + 0], lp[2 * half + 1],
                         lp[2 * half + 4], lp[2 * half + 5]);
        v16bf a1 = cvt16(lp[2 * half + 8], lp[2 * half + 9],
                         lp[2 * half + 12], lp[2 * half + 13]);

        // h = lat @ W1 : 8 N-tiles x 2 K-chunks, B frags = 2x b128 LDS loads
        v8f hc[8];
        #pragma unroll
        for (int t = 0; t < 8; ++t) {
            const bf16_t* b0p = w1_lds + t * 512 + fofs;          // chunk 0
            const bf16_t* b1p = w1_lds + (8 + t) * 512 + fofs;    // chunk 1
            v16bf b0 = mk16(*(const v8bf*)b0p, *(const v8bf*)(b0p + 8));
            v16bf b1 = mk16(*(const v8bf*)b1p, *(const v8bf*)(b1p + 8));
            v8f cz = {};
            cz = __builtin_amdgcn_wmma_f32_16x16x32_bf16(false, a0, false, b0, (short)0, cz, false, false);
            cz = __builtin_amdgcn_wmma_f32_16x16x32_bf16(false, a1, false, b1, (short)0, cz, false, false);
            hc[t] = cz;
        }

        // gelu in place + publish h to this wave's LDS slab
        #pragma unroll
        for (int t = 0; t < 8; ++t) {
            #pragma unroll
            for (int r = 0; r < 8; ++r) {
                const float v = gelu_sig(hc[t][r]);
                hc[t][r] = v;
                h_ws[wave][r + 8 * half][t * 16 + lm] = v;
            }
        }
        __syncthreads();

        // dot(h_row, x_proj_row): lane covers half the 128 columns of row lm
        float s = 0.f;
        const float* hr = &h_ws[wave][lm][half * 64];
        const float* xr = &xp_lds[lm][half * 64];
        #pragma unroll
        for (int f = 0; f < 64; f += 4) {
            float4 hv = *(const float4*)(hr + f);
            float4 xv = *(const float4*)(xr + f);
            s += hv.x * xv.x + hv.y * xv.y + hv.z * xv.z + hv.w * xv.w;
        }
        s += __shfl_xor(s, 16, 32);

        const float sc = scores[(size_t)(n0 + lm) * HK + g];
        const float av = gelu_sig(s) * sc;
        if (half == 0) act_ws[wave][lm] = av;
        __syncthreads();

        float am[8];
        #pragma unroll
        for (int r = 0; r < 8; ++r) am[r] = act_ws[wave][r + 8 * half];
        #pragma unroll
        for (int t = 0; t < 8; ++t)
            #pragma unroll
            for (int r = 0; r < 8; ++r)
                acc[t][r] += am[r] * hc[t][r];
    }

    // ---- reduce 4 per-wave accumulators to bf16 A matrix (x 1/num_heads) ----
    #pragma unroll
    for (int t = 0; t < 8; ++t)
        #pragma unroll
        for (int r = 0; r < 8; ++r)
            h_ws[wave][r + 8 * half][t * 16 + lm] = acc[t][r];
    __syncthreads();
    {
        const float* hf = &h_ws[0][0][0];
        for (int i = tid; i < TILE_M * D_HIDDEN; i += 128) {
            const float v = hf[i] + hf[2048 + i] + hf[4096 + i] + hf[6144 + i];
            accA[i] = (bf16_t)(0.25f * v);
        }
    }
    __syncthreads();

    // ---- Phase C: out[16][1024] = accA @ W_v ----
    {
        // warm L2 path for this wave's W_v fragment stream
        __builtin_prefetch(wvf + (size_t)(wave * 16) * 512, 0, 1);

        v16bf aC[4];
        #pragma unroll
        for (int c = 0; c < 4; ++c) {
            const bf16_t* ap = accA + lm * D_HIDDEN + c * 32 + 8 * half;
            aC[c] = mk16(*(const v8bf*)ap, *(const v8bf*)(ap + 16));
        }
        for (int j = 0; j < 16; ++j) {
            const int tg  = wave * 16 + j;
            const int col = tg * 16 + lm;
            v8f oc = {};
            #pragma unroll
            for (int c = 0; c < 4; ++c) {
                const bf16_t* bp = wvf + (size_t)(c * 64 + tg) * 512 + fofs;
                v16bf b = mk16(*(const v8bf*)bp, *(const v8bf*)(bp + 8));
                oc = __builtin_amdgcn_wmma_f32_16x16x32_bf16(false, aC[c], false, b, (short)0, oc, false, false);
            }
            #pragma unroll
            for (int r = 0; r < 8; ++r)
                out[(size_t)(n0 + r + 8 * half) * D_MODEL + col] = oc[r];
        }
    }
}

extern "C" void kernel_launch(void* const* d_in, const int* in_sizes, int n_in,
                              void* d_out, int out_size, void* d_ws, size_t ws_size,
                              hipStream_t stream) {
    const float* x      = (const float*)d_in[0];
    const float* scores = (const float*)d_in[1];
    const int*   eidx   = (const int*)  d_in[2];
    const float* lat    = (const float*)d_in[3];
    const float* W1     = (const float*)d_in[4];
    const float* Wu     = (const float*)d_in[5];
    const float* Wv     = (const float*)d_in[6];
    float* out = (float*)d_out;

    bf16_t* ws  = (bf16_t*)d_ws;                 // needs WS_ELEMS*2 = 540 KB scratch
    bf16_t* wuf = ws;
    bf16_t* wvf = ws + WU_ELEMS;
    bf16_t* w1f = ws + WU_ELEMS + WV_ELEMS;

    // one-time (per launch) weight convert + fragment swizzle
    convert_weights_kernel<<<(WS_ELEMS + 255) / 256, 256, 0, stream>>>(Wu, Wv, W1, ws);

    const int N = in_sizes[0] / D_MODEL;         // 8192 tokens
    fused_lowrank_moe_kernel<<<dim3(N / TILE_M), dim3(128), 0, stream>>>(
        x, scores, eidx, lat, wuf, wvf, w1f, out);
}
